// ARConv_65180423685385
// MI455X (gfx1250) — matrix operand order repaired and checked by reading, not compiled, not run on hardware
//
#include <hip/hip_runtime.h>
#include <hip/hip_bf16.h>
#include <math.h>

// ---------------------------------------------------------------------------
// CDNA5 (gfx1250) ARConv: NHWC-bf16 activations, direct-global WMMA GEMM.
//   A = weights (16 out-ch x 32 cin), pre-packed into exact per-lane image.
//   B = activations (32 cin x 16 px): per-lane fragment == contiguous 32B of
//       16 consecutive channels at one pixel (NHWC) -> plain b128 loads,
//       no LDS staging, no barriers in the conv hot loop.
//   V_WMMA_F32_16X16X32_BF16, fp32 accumulate.
// ---------------------------------------------------------------------------

typedef __attribute__((ext_vector_type(16))) __bf16 v16bf;
typedef __attribute__((ext_vector_type(8)))  float  v8f;

union Frag  { v16bf v; uint4 q[2]; };
union Pack8 { uint4 q; unsigned short u[8]; };

__device__ __forceinline__ unsigned short f2bf(float f) {
  unsigned int u = __float_as_uint(f);
  u += 0x7FFFu + ((u >> 16) & 1u);            // round-to-nearest-even
  return (unsigned short)(u >> 16);
}
__device__ __forceinline__ float bf2f(unsigned short u) {
  return __uint_as_float((unsigned int)u << 16);
}

#define Hh 128
#define Ww 128
#define Cc 64
#define Bb_ 8
#define HWsz (Hh * Ww)
#define PKN 36864   // ushorts per packed weight: 9 taps * 2 chunks * 4 ctiles * 32 lanes * 16

// ---------------------------------------------------------------------------
// NCHW fp32 -> NHWC bf16 (LDS tile transpose, coalesced in and out)
// ---------------------------------------------------------------------------
__global__ __launch_bounds__(256) void nchw_to_nhwc_bf16(
    const float* __restrict__ in, unsigned short* __restrict__ out) {
  __shared__ unsigned short T[32][66];
  const int w0 = blockIdx.x * 32, h = blockIdx.y, b = blockIdx.z;
  const int tid = threadIdx.x;
  #pragma unroll
  for (int i = 0; i < 8; ++i) {            // read: c-major, w fastest (coalesced)
    int idx = tid + i * 256;
    int c = idx >> 5, w = idx & 31;
    T[w][c] = f2bf(in[(((size_t)b * Cc + c) * Hh + h) * Ww + w0 + w]);
  }
  __syncthreads();
  #pragma unroll
  for (int i = 0; i < 8; ++i) {            // write: w-major, c fastest (coalesced)
    int idx = tid + i * 256;
    int w = idx >> 6, c = idx & 63;
    out[(((size_t)b * Hh + h) * Ww + w0 + w) * Cc + c] = T[w][c];
  }
}

// ---------------------------------------------------------------------------
// Weight repack: (O=64,Cin=64,3,3) fp32 -> per-lane A-fragment image (bf16).
// A (16x32 16-bit, contiguous K to match B): lane m=out-ch; VGPR v: K = 2v + 16*(lane/16).
// Flat index = (((t*2+cc)*4+ci)*32+lane)*16 + j.   wscale folds input scaling.
// ---------------------------------------------------------------------------
__global__ __launch_bounds__(256) void pack_w(const float* __restrict__ w,
                                              unsigned short* __restrict__ out,
                                              float wscale) {
  int idx = blockIdx.x * 256 + threadIdx.x;
  if (idx >= PKN) return;
  int j    = idx & 15;
  int lane = (idx >> 4) & 31;
  int ci   = (idx >> 9) & 3;
  int cc   = (idx >> 11) & 1;
  int t    = idx >> 12;
  int hl   = lane >> 4;
  int v    = j >> 1;
  int kb   = 2 * v + 16 * hl;
  int k    = kb + (j & 1);
  int cin  = cc * 32 + k;
  int o    = ci * 16 + (lane & 15);
  out[idx] = f2bf(w[(o * 64 + cin) * 9 + t] * wscale);
}

// ---------------------------------------------------------------------------
// 3x3 pad=1 conv 64->64 over NHWC bf16, WMMA, zero LDS / zero barriers.
// Block 256 thr = 8 waves; tile 32 px x 64 out-ch; wave (pi,ci) -> 16px x 16ch.
// MODE: 0=bias, 1=bias+lrelu, 2=bias+tanh, 3=bias+BN+lrelu
// ---------------------------------------------------------------------------
template<int MODE>
__global__ __launch_bounds__(256) void conv3x3_wmma(
    const unsigned short* __restrict__ xin, const unsigned short* __restrict__ wp,
    const float* __restrict__ bias,
    const float* __restrict__ bng, const float* __restrict__ bnb,
    const float* __restrict__ bnm, const float* __restrict__ bnv,
    unsigned short* __restrict__ out)
{
  const int tid = threadIdx.x, lane = tid & 31, wv = tid >> 5;
  const int ci = wv & 3, pi = wv >> 2;
  const int hl = lane >> 4, ln = lane & 15;
  const int wbase = blockIdx.x * 32, h = blockIdx.y, b = blockIdx.z;
  const int wpix = wbase + pi * 16 + ln;

  v8f acc = {};
  #pragma unroll
  for (int t = 0; t < 9; ++t) {
    const int dy = t / 3 - 1, dx = t % 3 - 1;
    const int hh = h + dy, wx = wpix + dx;
    const bool inb = (hh >= 0) && (hh < Hh) && (wx >= 0) && (wx < Ww);
    const size_t brow = inb ? (((size_t)b * Hh + hh) * Ww + wx) * Cc : 0;
    if (inb) __builtin_prefetch(xin + brow, 0, 3);   // global_prefetch_b8
    #pragma unroll
    for (int cc = 0; cc < 2; ++cc) {
      Frag a;                                         // packed weights (L2 hit)
      const uint4* ap = (const uint4*)(wp + (size_t)(((t * 2 + cc) * 4 + ci) * 32 + lane) * 16);
      a.q[0] = ap[0]; a.q[1] = ap[1];
      Frag bx;                                        // 32B contiguous channel run
      if (inb) {
        const uint4* bp = (const uint4*)(xin + brow + cc * 32 + hl * 16);
        bx.q[0] = bp[0]; bx.q[1] = bp[1];
      } else {
        bx.q[0] = make_uint4(0, 0, 0, 0); bx.q[1] = make_uint4(0, 0, 0, 0);
      }
      acc = __builtin_amdgcn_wmma_f32_16x16x32_bf16(false, a.v, false, bx.v,
                                                    (short)0, acc, false, false);
    }
  }

  // ---- epilogue: lane owns 8 consecutive out-channels of pixel wpix
  const int ch0 = ci * 16 + hl * 8;
  float z[8];
  #pragma unroll
  for (int i = 0; i < 8; ++i) z[i] = acc[i] + bias[ch0 + i];
  if (MODE == 3) {
    #pragma unroll
    for (int i = 0; i < 8; ++i) {
      float inv = bng[ch0 + i] * rsqrtf(bnv[ch0 + i] + 1e-5f);
      z[i] = (z[i] - bnm[ch0 + i]) * inv + bnb[ch0 + i];
    }
  }
  #pragma unroll
  for (int i = 0; i < 8; ++i) {
    if (MODE == 1 || MODE == 3) z[i] = (z[i] > 0.0f) ? z[i] : 0.01f * z[i];
    else if (MODE == 2)         z[i] = tanhf(z[i]);
  }
  Pack8 po;
  #pragma unroll
  for (int i = 0; i < 8; ++i) po.u[i] = f2bf(z[i]);
  *(uint4*)(out + ((((size_t)b * Hh + h) * Ww + wpix) * Cc + ch0)) = po.q;
}

// ---------------------------------------------------------------------------
// Scalar head over NHWC bf16 f: lrelu(BN(conv3x3 64->1)) -> sigmoid(BN(1x1))*8+1
// ---------------------------------------------------------------------------
__global__ __launch_bounds__(256) void head_k(
    const unsigned short* __restrict__ f,
    const float* __restrict__ w1, const float* __restrict__ b1,
    const float* __restrict__ g1, const float* __restrict__ be1,
    const float* __restrict__ m1, const float* __restrict__ v1,
    const float* __restrict__ w2, const float* __restrict__ b2,
    const float* __restrict__ g2, const float* __restrict__ be2,
    const float* __restrict__ m2, const float* __restrict__ v2,
    float* __restrict__ out)
{
  int p = blockIdx.x * 256 + threadIdx.x;
  if (p >= Bb_ * HWsz) return;
  int b = p >> 14, hw = p & (HWsz - 1);
  int h = hw >> 7, w = hw & 127;
  float acc = 0.0f;
  for (int t = 0; t < 9; ++t) {
    int hh = h + t / 3 - 1, ww = w + t % 3 - 1;
    if (hh < 0 || hh >= Hh || ww < 0 || ww >= Ww) continue;
    const unsigned short* fp = f + (((size_t)b * Hh + hh) * Ww + ww) * Cc;
    for (int c = 0; c < Cc; ++c) acc += bf2f(fp[c]) * w1[c * 9 + t];  // c contiguous
  }
  float inv1 = g1[0] * rsqrtf(v1[0] + 1e-5f);
  float z = (acc + b1[0] - m1[0]) * inv1 + be1[0];
  z = (z > 0.0f) ? z : 0.01f * z;
  float z2 = z * w2[0] + b2[0];
  float inv2 = g2[0] * rsqrtf(v2[0] + 1e-5f);
  z2 = (z2 - m2[0]) * inv2 + be2[0];
  float s = 1.0f / (1.0f + expf(-z2));
  out[p] = s * 8.0f + 1.0f;
}

// ---------------------------------------------------------------------------
// Deformable gather + stride-3 conv (GEMM K=9*64) + (.*m + b), fp32 NCHW out.
// ---------------------------------------------------------------------------
__global__ __launch_bounds__(256) void sample_conv_wmma(
    const unsigned short* __restrict__ xin, const float* __restrict__ Lm,
    const float* __restrict__ Wm, const unsigned short* __restrict__ wp,
    const float* __restrict__ kbias, const unsigned short* __restrict__ Mt,
    const unsigned short* __restrict__ Bt, float* __restrict__ out)
{
  __shared__ short qys[9][32];
  __shared__ short qxs[9][32];
  const int tid = threadIdx.x, lane = tid & 31, wv = tid >> 5;
  const int ci = wv & 3, pi = wv >> 2;
  const int hl = lane >> 4, ln = lane & 15;
  const int wbase = blockIdx.x * 32, h = blockIdx.y, b = blockIdx.z;

  // per-pixel coordinates (faithful torch .view interleave)
  if (tid < 32) {
    int m = tid, w = wbase + m;
    float L  = Lm[((size_t)b * Hh + h) * Ww + w];
    float Wd = Wm[((size_t)b * Hh + h) * Ww + w];
    float L3 = L * (1.0f / 3.0f), W3 = Wd * (1.0f / 3.0f);
    float chv[18];
    #pragma unroll
    for (int t = 0; t < 9; ++t) {
      chv[t]     = L3 * (float)(t / 3 - 1);
      chv[9 + t] = W3 * (float)(t % 3 - 1);
    }
    #pragma unroll
    for (int n = 0; n < 9; ++n) {
      float py = fminf(fmaxf((float)(h + 1) + chv[2 * n],     0.0f), 129.0f);
      float px = fminf(fmaxf((float)(w + 1) + chv[2 * n + 1], 0.0f), 129.0f);
      qys[n][m] = (short)floorf(py);
      qxs[n][m] = (short)floorf(px);
    }
  }
  __syncthreads();

  const int m = pi * 16 + ln;
  v8f acc = {};
  #pragma unroll
  for (int t = 0; t < 9; ++t) {
    const int qy = qys[t][m], qx = qxs[t][m];
    const bool inb = (qy >= 1) && (qy <= Hh) && (qx >= 1) && (qx <= Ww);
    const size_t brow = inb ? (((size_t)b * Hh + (qy - 1)) * Ww + (qx - 1)) * Cc : 0;
    #pragma unroll
    for (int cc = 0; cc < 2; ++cc) {
      Frag a;
      const uint4* ap = (const uint4*)(wp + (size_t)(((t * 2 + cc) * 4 + ci) * 32 + lane) * 16);
      a.q[0] = ap[0]; a.q[1] = ap[1];
      Frag bx;
      if (inb) {
        const uint4* bp = (const uint4*)(xin + brow + cc * 32 + hl * 16);
        bx.q[0] = bp[0]; bx.q[1] = bp[1];
      } else {
        bx.q[0] = make_uint4(0, 0, 0, 0); bx.q[1] = make_uint4(0, 0, 0, 0);
      }
      acc = __builtin_amdgcn_wmma_f32_16x16x32_bf16(false, a.v, false, bx.v,
                                                    (short)0, acc, false, false);
    }
  }

  // epilogue: z = acc + kb; out = z * m3 + b3  (fp32 NCHW final output)
  const int ch0  = ci * 16 + hl * 8;
  const int pixw = wbase + pi * 16 + ln;
  const size_t aoff = (((size_t)b * Hh + h) * Ww + pixw) * Cc + ch0;
  Pack8 pm, pb;
  pm.q = *(const uint4*)(Mt + aoff);
  pb.q = *(const uint4*)(Bt + aoff);
  #pragma unroll
  for (int i = 0; i < 8; ++i) {
    float z = acc[i] + kbias[ch0 + i];
    float r = z * bf2f(pm.u[i]) + bf2f(pb.u[i]);
    out[(((size_t)b * Cc + ch0 + i) * Hh + h) * Ww + pixw] = r;
  }
}

// ---------------------------------------------------------------------------
// Host orchestration.  Inputs: 0=x, then params depth-first (bn -> g,b,m,v).
// ---------------------------------------------------------------------------
extern "C" void kernel_launch(void* const* d_in, const int* in_sizes, int n_in,
                              void* d_out, int out_size, void* d_ws, size_t ws_size,
                              hipStream_t stream) {
  const float* P[64];
  for (int i = 0; i < n_in && i < 64; ++i) P[i] = (const float*)d_in[i];
  const float* x = P[0];

  const size_t S  = (size_t)Bb_ * Cc * HWsz;   // elements of one activation tensor
  const size_t SM = (size_t)Bb_ * HWsz;
  unsigned short* xbf = (unsigned short*)d_ws;
  unsigned short* A0  = xbf + S;
  unsigned short* A1  = A0 + S;
  unsigned short* Mb  = A1 + S;
  unsigned short* Bf  = Mb + S;
  float* Lmp = (float*)(Bf + S);
  float* Wmp = Lmp + SM;
  unsigned short* pk = (unsigned short*)(Wmp + SM);

  dim3 grid(Ww / 32, Hh, Bb_), blk(256);

  // x -> NHWC bf16 (once)
  nchw_to_nhwc_bf16<<<grid, blk, 0, stream>>>(x, xbf);

  // pack nine 64x64x3x3 weight sets (p1 weights absorb the x*100 scaling)
  const int   widx[9] = {1, 3, 5, 7, 9, 11, 13, 19, 49};
  const float wscl[9] = {1.f, 1.f, 1.f, 1.f, 1.f, 1.f, 100.f, 1.f, 1.f};
  for (int i = 0; i < 9; ++i)
    pack_w<<<144, 256, 0, stream>>>(P[widx[i]], pk + (size_t)i * PKN, wscl[i]);

  const float* dmy = P[0];
  // m branch
  conv3x3_wmma<1><<<grid, blk, 0, stream>>>(xbf, pk + 0 * (size_t)PKN, P[2],  dmy, dmy, dmy, dmy, A0);
  conv3x3_wmma<1><<<grid, blk, 0, stream>>>(A0,  pk + 1 * (size_t)PKN, P[4],  dmy, dmy, dmy, dmy, A1);
  conv3x3_wmma<2><<<grid, blk, 0, stream>>>(A1,  pk + 2 * (size_t)PKN, P[6],  dmy, dmy, dmy, dmy, Mb);
  // b branch
  conv3x3_wmma<1><<<grid, blk, 0, stream>>>(xbf, pk + 3 * (size_t)PKN, P[8],  dmy, dmy, dmy, dmy, A0);
  conv3x3_wmma<1><<<grid, blk, 0, stream>>>(A0,  pk + 4 * (size_t)PKN, P[10], dmy, dmy, dmy, dmy, A1);
  conv3x3_wmma<0><<<grid, blk, 0, stream>>>(A1,  pk + 5 * (size_t)PKN, P[12], dmy, dmy, dmy, dmy, Bf);
  // p branch (BN+lrelu epilogues; x*100 folded into packed weights)
  conv3x3_wmma<3><<<grid, blk, 0, stream>>>(xbf, pk + 6 * (size_t)PKN, P[14], P[15], P[16], P[17], P[18], A0);
  conv3x3_wmma<3><<<grid, blk, 0, stream>>>(A0,  pk + 7 * (size_t)PKN, P[20], P[21], P[22], P[23], P[24], A1);

  // heads -> l, w maps
  int hb = (int)((SM + 255) / 256);
  head_k<<<hb, 256, 0, stream>>>(A1, P[25], P[26], P[27], P[28], P[29], P[30],
                                 P[31], P[32], P[33], P[34], P[35], P[36], Lmp);
  head_k<<<hb, 256, 0, stream>>>(A1, P[37], P[38], P[39], P[40], P[41], P[42],
                                 P[43], P[44], P[45], P[46], P[47], P[48], Wmp);

  // gather + stride-3 conv + (.*m + b)
  sample_conv_wmma<<<grid, blk, 0, stream>>>(xbf, Lmp, Wmp, pk + 8 * (size_t)PKN,
                                             P[50], Mb, Bf, (float*)d_out);
}
